// HierarchicalMemory_44023414784311
// MI455X (gfx1250) — compile-verified, matrix operand order checked by reading, not run on hardware
//
#include <hip/hip_runtime.h>

// ---- types -----------------------------------------------------------------
typedef __attribute__((ext_vector_type(16))) __bf16       bf16x16;
typedef __attribute__((ext_vector_type(4)))  __bf16       bf16x4;
typedef __attribute__((ext_vector_type(8)))  float        f32x8;
typedef __attribute__((ext_vector_type(4)))  float        f32x4;
typedef __attribute__((ext_vector_type(4)))  unsigned int u32x4;

union ABFrag { u32x4 u[2]; bf16x16 v; };

#define P        400
#define PPAD     416      // 400 padded to 13 chunks of K=32
#define ROWS_LDS 352      // rows 0..339 needed after iter0; padded to 16
#define NT       512      // threads per block (16 waves)
#define NWAVE    16
#define NKC      13       // K chunks of 32
#define LAMBDA   0.9999f
#define ETA      0.5f
#define DECAY    0.8f

__device__ __forceinline__ float clip1(float x) { return fminf(1.f, fmaxf(-1.f, x)); }
__device__ __forceinline__ float leaky(float x) { return x >= 0.f ? x : 0.01f * x; }
// FREQS is strictly increasing -> mask(r,c) = (stream(c) >= stream(r))
__device__ __forceinline__ int streamOf(int i) {
  return (i < 100) ? 0 : (i < 200) ? 1 : (i < 280) ? 2 : (i < 340) ? 3 : 4;
}

__global__ __launch_bounds__(NT)
void tem_fused_kernel(const float* __restrict__ M_prev,
                      const float* __restrict__ p_inf,
                      const float* __restrict__ p_gen,
                      const float* __restrict__ query,
                      float* __restrict__ out_h,
                      float* __restrict__ out_M) {
  __shared__ __attribute__((aligned(16))) __bf16 sM[ROWS_LDS * PPAD]; // 286 KB bf16 M copy
  __shared__ __attribute__((aligned(16))) __bf16 sHbf[PPAD];          // bf16 shadow of h
  __shared__ float sH[P];
  __shared__ float sMv[P];
  __shared__ float sD[P];
  __shared__ float sSp[P];

  const int b    = blockIdx.x;
  const int tid  = threadIdx.x;
  const int lane = tid & 31;
  const int wave = tid >> 5;

  const float* Mp = M_prev + (size_t)b * (P * P);
  float*       Mo = out_M  + (size_t)b * (P * P);

  // ---- init: vectors + zero the K-padding (cols 400..415) -------------------
  for (int i = tid; i < P; i += NT) {
    float pi = p_inf[b * P + i];
    float pg = p_gen[b * P + i];
    sD[i]  = pi - pg;
    sSp[i] = pi + pg;
    sH[i]  = leaky(clip1(query[b * P + i]));   // h0
  }
  for (int i = tid; i < 340 * 8; i += NT) {    // 8 dwords of pad per row
    int r = i >> 3, dw = i & 7;
    ((unsigned int*)(sM + r * PPAD + P))[dw] = 0u;
  }
  if (tid < 8) ((unsigned int*)(sHbf + P))[tid] = 0u;
  __syncthreads();

  // ---- Phase A: Hebbian update (fp32, streamed) fused with iter-0 matvec ----
  // wave w owns rows [25w, 25w+25); lanes cover 128 columns per chunk (float4).
  for (int rr = 0; rr < 25; ++rr) {
    const int   r  = wave * 25 + rr;
    const float dr = sD[r];
    const int   sr = streamOf(r);
    float acc = 0.f;
    #pragma unroll
    for (int ch = 0; ch < 4; ++ch) {
      const int c = ch * 128 + lane * 4;
      if (c < P) {
        f32x4 m = __builtin_nontemporal_load((const f32x4*)(Mp + (size_t)r * P + c));
        f32x4 o;
        bf16x4 pk;
        #pragma unroll
        for (int j = 0; j < 4; ++j) {
          const int   cj = c + j;
          const float mk = (streamOf(cj) >= sr) ? ETA : 0.0f;
          float mn = clip1(fmaf(mk * dr, sSp[cj], LAMBDA * m[j]));
          o[j]  = mn;
          pk[j] = (__bf16)mn;
          acc   = fmaf(mn, sH[cj], acc);      // iter-0 matvec with h0
        }
        __builtin_nontemporal_store(o, (f32x4*)(Mo + (size_t)r * P + c));
        if (r < 340) *(bf16x4*)(sM + (size_t)r * PPAD + c) = pk;
      }
    }
    #pragma unroll
    for (int off = 16; off; off >>= 1) acc += __shfl_xor(acc, off, 32);
    if (lane == 0) sMv[r] = acc;
  }
  __syncthreads();

  // iter-0 h update (all streams active at it=0)
  for (int i = tid; i < P; i += NT) {
    float hn = leaky(clip1(fmaf(DECAY, sH[i], sMv[i])));
    sH[i]   = hn;
    sHbf[i] = (__bf16)hn;
  }
  __syncthreads();

  // ---- Phase B: iterations 1..4, LDS-resident bf16 matvec via WMMA ----------
  const int actives[4] = {340, 280, 200, 100};  // rows updated at it=1..4
  for (int it = 0; it < 4; ++it) {
    const int active = actives[it];
    const int ntiles = (active + 15) >> 4;
    const int g8     = (lane >> 4) * 8;        // K-group offset (ISA 16-bit A layout)
    const int m      = lane & 15;              // row within tile

    // B[k,n] = h[k] for all n: load all 13 broadcast fragments ONCE per
    // iteration, keep them in registers across the whole tile loop.
    ABFrag Bf[NKC];
    #pragma unroll
    for (int kc = 0; kc < NKC; ++kc) {
      Bf[kc].u[0] = *(const u32x4*)(sHbf + kc * 32 + g8);
      Bf[kc].u[1] = *(const u32x4*)(sHbf + kc * 32 + 16 + g8);
    }

    for (int t = wave; t < ntiles; t += NWAVE) {
      const int rowb = t << 4;
      const __bf16* arow = sM + (size_t)(rowb + m) * PPAD;

      // Prefetch ALL 13 A fragments (26 ds_load_b128, DScnt max 63 so legal),
      // then fence the scheduler so loads are not sunk back to their uses.
      ABFrag A[NKC];
      #pragma unroll
      for (int kc = 0; kc < NKC; ++kc) {
        A[kc].u[0] = *(const u32x4*)(arow + kc * 32 + g8);
        A[kc].u[1] = *(const u32x4*)(arow + kc * 32 + 16 + g8);
      }
      __builtin_amdgcn_sched_barrier(0);       // hard fence: loads stay ahead of WMMAs

      f32x8 c = {0.f, 0.f, 0.f, 0.f, 0.f, 0.f, 0.f, 0.f};
      #pragma unroll
      for (int kc = 0; kc < NKC; ++kc) {
        c = __builtin_amdgcn_wmma_f32_16x16x32_bf16(
                false, A[kc].v, false, Bf[kc].v, (short)0, c, false, false);
      }
      // D replicated across N; C layout: lane<16 -> M=i, lane>=16 -> M=i+8
      if (lane == 0) {
        #pragma unroll
        for (int i = 0; i < 8; ++i) sMv[rowb + i] = c[i];
      }
      if (lane == 16) {
        #pragma unroll
        for (int i = 0; i < 8; ++i) sMv[rowb + 8 + i] = c[i];
      }
    }
    __syncthreads();
    for (int i = tid; i < active; i += NT) {
      float hn = leaky(clip1(fmaf(DECAY, sH[i], sMv[i])));
      sH[i]   = hn;
      sHbf[i] = (__bf16)hn;
    }
    __syncthreads();
  }

  // ---- write h ---------------------------------------------------------------
  for (int i = tid; i < P; i += NT) out_h[b * P + i] = sH[i];
}

extern "C" void kernel_launch(void* const* d_in, const int* in_sizes, int n_in,
                              void* d_out, int out_size, void* d_ws, size_t ws_size,
                              hipStream_t stream) {
  (void)in_sizes; (void)n_in; (void)out_size; (void)d_ws; (void)ws_size;
  const float* M_prev = (const float*)d_in[0];
  const float* p_inf  = (const float*)d_in[1];
  const float* p_gen  = (const float*)d_in[2];
  const float* query  = (const float*)d_in[3];
  float* out   = (float*)d_out;
  float* out_h = out;                    // [512, 400]   (first return value)
  float* out_M = out + (size_t)512 * P;  // [512,400,400] (second return value)
  tem_fused_kernel<<<dim3(512), dim3(NT), 0, stream>>>(
      M_prev, p_inf, p_gen, query, out_h, out_M);
}
